// ScannedRNN_4801773437641
// MI455X (gfx1250) — matrix operand order repaired and checked by reading, not compiled, not run on hardware
//
#include <hip/hip_runtime.h>

// ---- problem constants (T,B,H) = (512,512,128), G = 3H = 384 ----
#define T_STEPS 512
#define BATCH   512
#define HDIM    128
#define GDIM    384
#define ROWS    16      // batch rows per workgroup (WMMA M)
#define NTILES  24      // GDIM / 16
#define KBLK    4       // HDIM / 32 (K per bf16 WMMA = 32)
#define THREADS 256     // 8 wave32s
#define NWGS    (BATCH / ROWS)   // 32

typedef __attribute__((ext_vector_type(16))) __bf16 v16bf;
typedef __attribute__((ext_vector_type(8)))  __bf16 v8bf;
typedef __attribute__((ext_vector_type(8)))  float  v8f;

// ---- LDS layout (bytes). 258 KB total -- needs CDNA5's 320 KB/WGP. ----
#define WI_OFF   0            // Wi bf16, B-fragment order: 24*4*32 frags * 32B = 96 KB
#define WH_OFF   98304        // Wh bf16, same layout (read once into VGPRs)  = 96 KB
#define XB_OFF   196608       // x tile bf16   16x128                        = 4 KB
#define HB_OFF   200704       // h carry bf16  16x128                        = 4 KB
#define HF_OFF   204800       // h carry f32   16x128                        = 8 KB
#define XG_OFF   212992       // xg f32        16x384                        = 24 KB
#define HG_OFF   237568       // hg f32        16x384                        = 24 KB
#define BI_OFF   262144       // bi f32 384                                  = 1.5 KB
#define BHN_OFF  263680       // bhn f32 128                                 = 0.5 KB
#define LDS_BYTES 264192

__device__ __forceinline__ __bf16 f2bf(float f) { return (__bf16)f; }

__device__ __forceinline__ float sigmoidf(float x) {
  return 1.0f / (1.0f + __expf(-x));
}
// tanh(x) = 2*sigmoid(2x) - 1 : one v_exp_f32 instead of the ocml tanh path
__device__ __forceinline__ float tanh_fast(float x) {
  return 2.0f / (1.0f + __expf(-2.0f * x)) - 1.0f;
}

// LDS-only barrier: drains DScnt only (not LOADcnt/STOREcnt), so global
// prefetches of x[t+1] and the ys stores stay outstanding across it.
// Wait-before-signal: each wave's LDS ops are complete before it signals,
// hence visible to every wave that leaves the barrier.
__device__ __forceinline__ void lds_barrier() {
  asm volatile(
      "s_wait_dscnt 0x0\n\t"
      "s_barrier_signal -1\n\t"
      "s_barrier_wait -1" ::: "memory");
}

// A-matrix fragment (16x32 bf16) from a row-major 16x128 bf16 LDS tile.
// ISA layout: lanes 0-15 row M=lane, K = {kb*32+0..7, kb*32+16..23};
//             lanes 16-31 row M=lane-16, K = {kb*32+8..15, kb*32+24..31}.
__device__ __forceinline__ v16bf load_afrag(const __bf16* src, int lane, int kb) {
  int row   = lane & 15;
  int kbase = kb * 32 + ((lane & 16) ? 8 : 0);
  const __bf16* p = src + row * HDIM + kbase;
  v8bf lo = *(const v8bf*)(p);        // ds_load_b128
  v8bf hi = *(const v8bf*)(p + 16);   // ds_load_b128
  v16bf a;
#pragma unroll
  for (int i = 0; i < 8; ++i) { a[i] = lo[i]; a[8 + i] = hi[i]; }
  return a;
}

__global__ __launch_bounds__(THREADS, 1)
void gru_scan_kernel(const float* __restrict__ x,
                     const unsigned char* __restrict__ resets,  // jnp.bool_ -> 1 byte
                     const float* __restrict__ h0,
                     const float* __restrict__ Wi,
                     const float* __restrict__ bi,
                     const float* __restrict__ Wh,
                     const float* __restrict__ bhn,
                     float* __restrict__ out) {
  extern __shared__ char smem[];
  __bf16* wiF = (__bf16*)(smem + WI_OFF);
  __bf16* whF = (__bf16*)(smem + WH_OFF);
  __bf16* xB  = (__bf16*)(smem + XB_OFF);
  __bf16* hB  = (__bf16*)(smem + HB_OFF);
  float*  hF  = (float*)(smem + HF_OFF);
  float*  xG  = (float*)(smem + XG_OFF);
  float*  hG  = (float*)(smem + HG_OFF);
  float*  biL = (float*)(smem + BI_OFF);
  float*  bnL = (float*)(smem + BHN_OFF);

  const int tid  = threadIdx.x;
  const int lane = tid & 31;
  const int w    = tid >> 5;           // wave id 0..7, owns N-cols [w*48, w*48+48)
  const int b0   = blockIdx.x * ROWS;

  // Per-thread ownership for stage & gate phases: one row, 8 contiguous channels.
  const int srow = tid >> 4;           // 0..15
  const int sc0  = (tid & 15) * 8;     // 0,8,...,120

  // ---- one-time: stage weights into WMMA B-fragment order (bf16) ----
  // B 32x16 layout: lanes 0-15 hold K=kb*32+{0..7,16..23} of col N=lane (2/VGPR);
  //                 lanes 16-31 hold K=kb*32+{8..15,24..31}.
  for (int idx = tid; idx < NTILES * KBLK * 32 * 16; idx += THREADS) {
    int e    = idx & 15;          // element within lane's 16 bf16
    int ln   = (idx >> 4) & 31;   // lane
    int kb   = (idx >> 9) & 3;    // k-block
    int tile = idx >> 11;         // N-tile
    int k = kb * 32 + ((ln & 16) ? 8 : 0) + (e & 7) + ((e & 8) ? 16 : 0);
    int n = tile * 16 + (ln & 15);
    wiF[idx] = f2bf(Wi[k * GDIM + n]);
    whF[idx] = f2bf(Wh[k * GDIM + n]);
  }
  for (int i = tid; i < GDIM; i += THREADS) biL[i] = bi[i];
  for (int i = tid; i < HDIM; i += THREADS) bnL[i] = bhn[i];
  {
    // carry init: same ownership as stage/gate phases
    v8f h8 = *(const v8f*)(h0 + (b0 + srow) * HDIM + sc0);
    *(v8f*)(hF + srow * HDIM + sc0) = h8;
    *(v8bf*)(hB + srow * HDIM + sc0) = __builtin_convertvector(h8, v8bf);
  }
  __syncthreads();   // one-time full barrier is fine here

  // ---- hoist loop-invariant Wh B-fragments into registers (96 VGPRs) ----
  v16bf bwh[3][KBLK];
#pragma unroll
  for (int tt = 0; tt < 3; ++tt)
#pragma unroll
    for (int kb = 0; kb < KBLK; ++kb)
      bwh[tt][kb] =
          *(const v16bf*)(whF + (((w * 3 + tt) * KBLK + kb) * 32 + lane) * 16);

  // ---- software pipeline: keep next step's x chunk + reset flag in registers ----
  const long           xstride = (long)BATCH * HDIM;
  const float*         xp = x + (long)(b0 + srow) * HDIM + sc0;
  const unsigned char* rp = resets + (b0 + srow);
  float*               op = out + (long)(b0 + srow) * HDIM + sc0;

  v8f           xv = *(const v8f*)xp;   // x[0] chunk
  unsigned char rv = *rp;               // resets[0]

  for (int t = 0; t < T_STEPS; ++t) {
    // ---- stage x[t] tile as bf16; mask carry where reset (same-thread h ownership) ----
    *(v8bf*)(xB + srow * HDIM + sc0) = __builtin_convertvector(xv, v8bf);
    if (rv) {
      v8f zf = {}; v8bf zb = {};
      *(v8f*)(hF + srow * HDIM + sc0) = zf;
      *(v8bf*)(hB + srow * HDIM + sc0) = zb;
    }
    // prefetch next step's x chunk + reset flag; the LDS-only barriers below do
    // NOT drain LOADcnt, so these stay in flight across GEMM + gate phases.
    if (t + 1 < T_STEPS) {
      xp += xstride; rp += BATCH;
      xv = *(const v8f*)xp;
      rv = *rp;
    }
    lds_barrier();   // xB/hB visible to all waves' A-fragment loads

    // ---- fused dual GEMM: xg = x@Wi (Wi streamed from LDS), hg = h@Wh (regs) ----
    v8f accX[3] = {{}, {}, {}};
    v8f accH[3] = {{}, {}, {}};
#pragma unroll
    for (int kb = 0; kb < KBLK; ++kb) {
      v16bf ax = load_afrag(xB, lane, kb);
      v16bf ah = load_afrag(hB, lane, kb);
#pragma unroll
      for (int tt = 0; tt < 3; ++tt) {
        const v16bf bwi =
            *(const v16bf*)(wiF + (((w * 3 + tt) * KBLK + kb) * 32 + lane) * 16);
        accX[tt] = __builtin_amdgcn_wmma_f32_16x16x32_bf16(false, ax, false, bwi,
                                                           (short)0, accX[tt],
                                                           false, false);
        accH[tt] = __builtin_amdgcn_wmma_f32_16x16x32_bf16(false, ah, false,
                                                           bwh[tt][kb], (short)0,
                                                           accH[tt], false, false);
      }
    }
    // C/D layout: VGPR i -> M=i (lanes 0-15) / M=8+i (lanes 16-31)
    {
      int mofs = (lane & 16) ? 8 : 0;
#pragma unroll
      for (int tt = 0; tt < 3; ++tt) {
        int n = (w * 3 + tt) * 16 + (lane & 15);
#pragma unroll
        for (int i = 0; i < 8; ++i) {
          xG[(mofs + i) * GDIM + n] = accX[tt][i];
          hG[(mofs + i) * GDIM + n] = accH[tt][i];
        }
      }
    }
    lds_barrier();   // xG/hG complete before gate math

    // ---- gate math, carry update, output store (vectorized, same-thread h ownership) ----
    {
      const float* xgp = xG + srow * GDIM;
      const float* hgp = hG + srow * GDIM;
      v8f xr = *(const v8f*)(xgp + sc0);
      v8f xz = *(const v8f*)(xgp + 128 + sc0);
      v8f xn = *(const v8f*)(xgp + 256 + sc0);
      v8f hr = *(const v8f*)(hgp + sc0);
      v8f hz = *(const v8f*)(hgp + 128 + sc0);
      v8f hn = *(const v8f*)(hgp + 256 + sc0);
      v8f bir = *(const v8f*)(biL + sc0);
      v8f biz = *(const v8f*)(biL + 128 + sc0);
      v8f bin = *(const v8f*)(biL + 256 + sc0);
      v8f bn8 = *(const v8f*)(bnL + sc0);
      v8f hold = *(const v8f*)(hF + srow * HDIM + sc0);
      v8f hnew;
#pragma unroll
      for (int i = 0; i < 8; ++i) {
        float r  = sigmoidf(xr[i] + bir[i] + hr[i]);
        float z  = sigmoidf(xz[i] + biz[i] + hz[i]);
        float nn = tanh_fast(xn[i] + bin[i] + r * (hn[i] + bn8[i]));
        hnew[i] = (1.0f - z) * nn + z * hold[i];
      }
      *(v8f*)(hF + srow * HDIM + sc0) = hnew;
      *(v8bf*)(hB + srow * HDIM + sc0) = __builtin_convertvector(hnew, v8bf);
      *(v8f*)op = hnew;   // coalesced 32B store; never drained by lds_barrier
      op += xstride;
    }
    // No end-of-step barrier: stage(t+1) touches only this thread's hF/hB elements;
    // the stage barrier orders xB/hB writes vs. other waves' A-fragment reads; the
    // acc-store barrier orders xG/hG reuse vs. gate reads (wait-then-signal).
  }
}

extern "C" void kernel_launch(void* const* d_in, const int* in_sizes, int n_in,
                              void* d_out, int out_size, void* d_ws, size_t ws_size,
                              hipStream_t stream) {
  (void)in_sizes; (void)n_in; (void)out_size; (void)d_ws; (void)ws_size;
  const float*         x      = (const float*)d_in[0];
  const unsigned char* resets = (const unsigned char*)d_in[1];  // jnp.bool_
  const float*         h0     = (const float*)d_in[2];
  const float*         Wi     = (const float*)d_in[3];
  const float*         bi     = (const float*)d_in[4];
  const float*         Wh     = (const float*)d_in[5];
  const float*         bhn    = (const float*)d_in[6];
  float*               out    = (float*)d_out;

  gru_scan_kernel<<<dim3(NWGS), dim3(THREADS), LDS_BYTES, stream>>>(
      x, resets, h0, Wi, bi, Wh, bhn, out);
}
